// TwoInputGeometricBilinear_13537736917624
// MI455X (gfx1250) — compile-verified, matrix operand order checked by reading, not run on hardware
//
#include <hip/hip_runtime.h>
#include <hip/hip_bf16.h>

// ---------------------------------------------------------------------------
// Types for WMMA (CDNA5 / gfx1250, wave32)
// ---------------------------------------------------------------------------
typedef __attribute__((ext_vector_type(16))) __bf16          v16bf;
typedef __attribute__((ext_vector_type(16))) unsigned short  v16u;
typedef __attribute__((ext_vector_type(8)))  float           v8f;

// ---------------------------------------------------------------------------
// Compile-time PGA(3,0,0,1) tables (mirrors the reference _build_tables).
// Blades as 4-bit masks; bit 0 = degenerate generator e0 (e0^2 = 0).
// All tables below are used ONLY in constexpr / template contexts.
// ---------------------------------------------------------------------------
struct Term { unsigned char i, j, k; signed char s; };

constexpr int BLADE_MASK[16] = {0,1,2,4,8,3,5,9,6,10,12,7,11,13,14,15};
constexpr int GRADE_C[16]    = {0,1,1,1,1,2,2,2,2,2,2,3,3,3,3,4};
// output comp i = e0 ^ blade j  ->  source component j (sign is +1), else -1
constexpr int E0SRC_C[16]    = {-1,0,-1,-1,-1,2,3,4,-1,-1,-1,8,9,10,-1,14};

constexpr int maskToIdx(int m) {
  for (int i = 0; i < 16; ++i) if (BLADE_MASK[i] == m) return i;
  return -1;
}
constexpr int popc4(unsigned x) { int c = 0; for (int b = 0; b < 4; ++b) c += (x >> b) & 1; return c; }
// parity of bubble-sorting concat(list(a), list(b)) -> sign
constexpr int reorderSign(unsigned a, unsigned b) {
  int sw = 0;
  for (int g = 0; g < 4; ++g)
    if ((b >> g) & 1) sw += popc4(a >> (g + 1));  // elems of a strictly greater than g
  return (sw & 1) ? -1 : 1;
}

struct CayleyTab { Term t[192]; };
constexpr CayleyTab makeCayley() {
  CayleyTab T{};
  int n = 0;
  for (int j = 0; j < 16; ++j)
    for (int k = 0; k < 16; ++k) {
      unsigned am = (unsigned)BLADE_MASK[j], bm = (unsigned)BLADE_MASK[k];
      if (am & bm & 1u) continue;                      // shared e0 -> 0 (degenerate metric)
      T.t[n++] = Term{(unsigned char)maskToIdx((int)(am ^ bm)),
                      (unsigned char)j, (unsigned char)k,
                      (signed char)reorderSign(am, bm)};
    }
  return T;
}
struct JoinTab { Term t[81]; };
constexpr JoinTab makeJoin() {
  JoinTab T{};
  int sg[16] = {};
  for (int j = 0; j < 16; ++j)
    sg[j] = reorderSign((unsigned)BLADE_MASK[j], 15u ^ (unsigned)BLADE_MASK[j]);
  int n = 0;
  for (int j = 0; j < 16; ++j)
    for (int k = 0; k < 16; ++k) {
      unsigned a = 15u ^ (unsigned)BLADE_MASK[j], b = 15u ^ (unsigned)BLADE_MASK[k];
      if (a & b) continue;                             // wedge of complements
      int w  = reorderSign(a, b);
      int mp = maskToIdx((int)(15u ^ (a | b)));        // complement of wedge result
      T.t[n++] = Term{(unsigned char)mp, (unsigned char)j, (unsigned char)k,
                      (signed char)(sg[j] * sg[k] * w * sg[mp])};
    }
  return T;
}
constexpr CayleyTab CAYLEY = makeCayley();
constexpr JoinTab   JOIN   = makeJoin();

// ---------------------------------------------------------------------------
// Workspace layout: bf16 WMMA B-fragments, 512 u16 per fragment (32 lanes x 16)
// ---------------------------------------------------------------------------
constexpr int FRAG        = 512;
constexpr int FR_SIDE     = 36;                 // 9 grade-mats * 2 n-tiles * 2 k-steps
constexpr int OFF_WOUT_F  = 4 * FR_SIDE;        // 144: out equi weights (9*4*2 = 72 frags)
constexpr int OFF_S2MV_F  = OFF_WOUT_F + 72;    // 216: 4 side s2mv sets, 4 frags each
constexpr int OFF_S2MVO_F = OFF_S2MV_F + 16;    // 232: s2mv_out (4 nt * 4 kb)
constexpr int OFF_MV2S_F  = OFF_S2MVO_F + 16;   // 248: mv2s_out (4 nt * 2 kb)
constexpr int OFF_S2S_F   = OFF_MV2S_F + 8;     // 256: s2s_out (4 nt * 4 kb)
constexpr int TOTAL_FRAG  = OFF_S2S_F + 16;     // 272
constexpr int WS_U16      = TOTAL_FRAG * FRAG;  // 139264 u16 = 278528 bytes of d_ws

// ---------------------------------------------------------------------------
// Helpers
// ---------------------------------------------------------------------------
__device__ __forceinline__ unsigned short f2bf(float f) {
  unsigned int u = __float_as_uint(f);
  u += 0x7fffu + ((u >> 16) & 1u);               // round-to-nearest-even
  return (unsigned short)(u >> 16);
}
__device__ __forceinline__ float bf2f(unsigned short h) {
  return __uint_as_float(((unsigned int)h) << 16);
}
// C = A(16x32 bf16) * B(32x16 bf16) + C   (f32 accumulate)
__device__ __forceinline__ v8f wmma_bf16_acc(const unsigned short* a,
                                             const unsigned short* b, v8f c) {
  v16u au = *(const v16u*)a;
  v16u bu = *(const v16u*)b;
  return __builtin_amdgcn_wmma_f32_16x16x32_bf16(
      false, __builtin_bit_cast(v16bf, au),
      false, __builtin_bit_cast(v16bf, bu),
      (short)0, c, false, false);
}
// Template-unrolled sparse bilinear accumulation -> all indices are constants
template <int N, int CNT>
__device__ __forceinline__ void gpTerms(float* acc, const float* L, const float* R) {
  if constexpr (N < CNT) {
    constexpr Term tm = CAYLEY.t[N];
    acc[tm.i] += (float)tm.s * L[tm.j] * R[tm.k];
    gpTerms<N + 1, CNT>(acc, L, R);
  }
}
template <int N, int CNT>
__device__ __forceinline__ void joinTerms(float* acc, const float* L, const float* R) {
  if constexpr (N < CNT) {
    constexpr Term tm = JOIN.t[N];
    acc[tm.i] += (float)tm.s * L[tm.j] * R[tm.k];
    joinTerms<N + 1, CNT>(acc, L, R);
  }
}

// ---------------------------------------------------------------------------
// Templated equi-linear component GEMMs: COMP is compile-time, so grade /
// e0-source lookups are constexpr, all fragment offsets fold to literals,
// there is no runtime branching (EXEC stays all-ones around every WMMA), and
// B-fragments shared between same-grade components get CSE'd by the compiler.
//
// aB = astage + lane*16          (A fragments, offset (comp*2+kb)*512)
// wB = ws + (wfrag0 + nt*2)*FRAG + lane*16   (side: +(g*4+kb)*FRAG; out: +(g*8+kb)*FRAG)
// sA = sstage + skb0*512 + lane*16
// sW = ws + (sfrag0 + nt*?)*FRAG + lane*16
// ---------------------------------------------------------------------------
template <int COMP>
__device__ __forceinline__ void side_comp(const unsigned short* aB, const unsigned short* wB,
                                          const unsigned short* sA, const unsigned short* sW,
                                          unsigned short* qPtr) {
  constexpr int g = GRADE_C[COMP];
  v8f c = {0.f, 0.f, 0.f, 0.f, 0.f, 0.f, 0.f, 0.f};
  c = wmma_bf16_acc(aB + (COMP * 2 + 0) * 512, wB + (g * 4 + 0) * FRAG, c);
  c = wmma_bf16_acc(aB + (COMP * 2 + 1) * 512, wB + (g * 4 + 1) * FRAG, c);
  if constexpr (E0SRC_C[COMP] >= 0) {
    constexpr int j  = E0SRC_C[COMP];
    constexpr int a5 = 5 + GRADE_C[j];
    c = wmma_bf16_acc(aB + (j * 2 + 0) * 512, wB + (a5 * 4 + 0) * FRAG, c);
    c = wmma_bf16_acc(aB + (j * 2 + 1) * 512, wB + (a5 * 4 + 1) * FRAG, c);
  }
  if constexpr (COMP == 0) {
    c = wmma_bf16_acc(sA + 0 * 512, sW + 0 * FRAG, c);
    c = wmma_bf16_acc(sA + 1 * 512, sW + 1 * FRAG, c);
  }
#pragma unroll
  for (int r = 0; r < 8; ++r) qPtr[r * 512] = f2bf(c[r]);  // quad[t][y][COMP], t-stride 512
}
template <int COMP = 0>
__device__ __forceinline__ void side_all(const unsigned short* aB, const unsigned short* wB,
                                         const unsigned short* sA, const unsigned short* sW,
                                         unsigned short* qBase) {
  if constexpr (COMP < 16) {
    side_comp<COMP>(aB, wB, sA, sW, qBase + COMP);
    side_all<COMP + 1>(aB, wB, sA, sW, qBase);
  }
}

template <int COMP>
__device__ __forceinline__ void out_comp(const unsigned short* aB, const unsigned short* wB,
                                         const unsigned short* sA, const unsigned short* sW,
                                         float* oPtr) {
  constexpr int g = GRADE_C[COMP];
  v8f c = {0.f, 0.f, 0.f, 0.f, 0.f, 0.f, 0.f, 0.f};
  c = wmma_bf16_acc(aB + (COMP * 2 + 0) * 512, wB + (g * 8 + 0) * FRAG, c);
  c = wmma_bf16_acc(aB + (COMP * 2 + 1) * 512, wB + (g * 8 + 1) * FRAG, c);
  if constexpr (E0SRC_C[COMP] >= 0) {
    constexpr int j  = E0SRC_C[COMP];
    constexpr int a5 = 5 + GRADE_C[j];
    c = wmma_bf16_acc(aB + (j * 2 + 0) * 512, wB + (a5 * 8 + 0) * FRAG, c);
    c = wmma_bf16_acc(aB + (j * 2 + 1) * 512, wB + (a5 * 8 + 1) * FRAG, c);
  }
  if constexpr (COMP == 0) {
#pragma unroll
    for (int skb = 0; skb < 4; ++skb)
      c = wmma_bf16_acc(sA + skb * 512, sW + skb * FRAG, c);
  }
#pragma unroll
  for (int r = 0; r < 8; ++r) oPtr[r * 1024] = c[r];       // otile[t][y][COMP], t-stride 1024
}
template <int COMP = 0>
__device__ __forceinline__ void out_all(const unsigned short* aB, const unsigned short* wB,
                                        const unsigned short* sA, const unsigned short* sW,
                                        float* oBase) {
  if constexpr (COMP < 16) {
    out_comp<COMP>(aB, wB, sA, sW, oBase + COMP);
    out_all<COMP + 1>(aB, wB, sA, sW, oBase);
  }
}

// ---------------------------------------------------------------------------
// Kernel 1: expand all weights into bf16 WMMA-B-native fragments in workspace
// B layout (mirror of the documented A layout): lane = 16*h + n,
// slot s holds K = (s&7) + 8*h + 16*(s>>3); element = B[k][n] = W[n][k].
// ---------------------------------------------------------------------------
__global__ void expand_weights(
    const float* __restrict__ wL,  const float* __restrict__ wJL,
    const float* __restrict__ wR,  const float* __restrict__ wJR,
    const float* __restrict__ sL,  const float* __restrict__ sJL,
    const float* __restrict__ sR,  const float* __restrict__ sJR,
    const float* __restrict__ wO,  const float* __restrict__ sO,
    const float* __restrict__ mv2s, const float* __restrict__ s2s,
    unsigned short* __restrict__ ws) {
  const int id = blockIdx.x * blockDim.x + threadIdx.x;
  if (id >= WS_U16) return;
  const int slot = id & 15;
  const int lane = (id >> 4) & 31;
  const int frag = id >> 9;
  const int h = lane >> 4, n16 = lane & 15;
  const int K = (slot & 7) + 8 * h + 16 * (slot >> 3);

  float val;
  if (frag < 144) {                                  // 4 side equi weight sets
    const int e = frag / FR_SIDE, r = frag % FR_SIDE;
    const int a = r >> 2, nt = (r >> 1) & 1, kb = r & 1;
    const int y = nt * 16 + n16, x = kb * 32 + K;
    const float* w = (e == 0) ? wL : (e == 1) ? wJL : (e == 2) ? wR : wJR;
    val = w[(y * 64 + x) * 9 + a];
  } else if (frag < OFF_S2MV_F) {                    // output equi weights
    const int r = frag - OFF_WOUT_F;
    const int a = r >> 3, nt = (r >> 1) & 3, kb = r & 1;
    const int y = nt * 16 + n16, x = kb * 32 + K;
    val = wO[(y * 64 + x) * 9 + a];
  } else if (frag < OFF_S2MVO_F) {                   // side s2mv
    const int r = frag - OFF_S2MV_F;
    const int e = r >> 2, nt = (r >> 1) & 1, skb = r & 1;
    const int y = nt * 16 + n16, s = skb * 32 + K;
    const float* sp = (e == 0) ? sL : (e == 1) ? sJL : (e == 2) ? sR : sJR;
    val = sp[y * 64 + s];
  } else if (frag < OFF_MV2S_F) {                    // s2mv_out (64 x 128)
    const int r = frag - OFF_S2MVO_F;
    const int nt = r >> 2, skb = r & 3;
    val = sO[(nt * 16 + n16) * 128 + skb * 32 + K];
  } else if (frag < OFF_S2S_F) {                     // mv2s_out: B[x][o] = mv2s[o][x]
    const int r = frag - OFF_MV2S_F;
    const int nt = r >> 1, kb = r & 1;
    val = mv2s[(nt * 16 + n16) * 64 + kb * 32 + K];
  } else {                                           // s2s_out (64 x 128)
    const int r = frag - OFF_S2S_F;
    const int nt = r >> 2, skb = r & 3;
    val = s2s[(nt * 16 + n16) * 128 + skb * 32 + K];
  }
  ws[id] = f2bf(val);
}

// ---------------------------------------------------------------------------
// Kernel 2: main fused pipeline. 1 workgroup (2 waves) per 16-token tile.
// Wave w owns n-tile(s): side equis nt=w, out equi / out_s nt in {2w, 2w+1}.
// ---------------------------------------------------------------------------
__global__ __launch_bounds__(64) void ga_bilinear_kernel(
    const float* __restrict__ mv1, const float* __restrict__ mv2,
    const float* __restrict__ s1,  const float* __restrict__ s2,
    const float* __restrict__ refmv,
    const unsigned short* __restrict__ ws,
    const float* __restrict__ bias,
    float* __restrict__ out_mv, float* __restrict__ out_s) {
  __attribute__((aligned(32))) __shared__ unsigned short astage[16 * 2 * 32 * 16]; // 32 KB
  __attribute__((aligned(32))) __shared__ unsigned short sstage[4 * 32 * 16];      //  4 KB
  __attribute__((aligned(32))) __shared__ unsigned short quads[4 * 16 * 32 * 16];  // 64 KB

  const int tid  = threadIdx.x;
  const int wave = tid >> 5;
  const int lane = tid & 31;
  const int tile = blockIdx.x;
  const int tok0 = tile * 16;
  const int h    = lane >> 4, n16 = lane & 15;

  const unsigned short* aB = astage + lane * 16;

  // stage a 16-token multivector tile into WMMA-A-native per-component layout
  auto stage_mv = [&](const float* __restrict__ src) {
    for (int base = tid * 4; base < 16384; base += 256) {
      const float4 v = *(const float4*)(src + base);
      const float vv[4] = {v.x, v.y, v.z, v.w};
      const int t  = base >> 10;
      const int x  = (base >> 4) & 63;
      const int i0 = base & 15;                      // comps i0..i0+3, i0 in {0,4,8,12}
      const int kb = x >> 5, K = x & 31;
      const int slot = (K & 7) + 8 * (K >> 4);
      const int ln   = ((K >> 3) & 1) * 16 + t;
#pragma unroll
      for (int q = 0; q < 4; ++q)
        astage[(((i0 + q) * 2 + kb) * 32 + ln) * 16 + slot] = f2bf(vv[q]);
    }
  };
  auto stage_s = [&](const float* __restrict__ src, int skb0) {
    for (int idx = tid; idx < 1024; idx += 64) {
      const int t = idx >> 6, s = idx & 63;
      const int kb = s >> 5, K = s & 31;
      const int slot = (K & 7) + 8 * (K >> 4);
      const int ln   = ((K >> 3) & 1) * 16 + t;
      sstage[((skb0 + kb) * 32 + ln) * 16 + slot] = f2bf(src[idx]);
    }
  };
  auto run_side = [&](int wfrag0, int sfrag0, int skb0, unsigned short* qdst) {
    const int nt = wave;                             // wave owns one n-tile
    const unsigned short* wB = ws + (wfrag0 + nt * 2) * FRAG + lane * 16;
    const unsigned short* sW = ws + (sfrag0 + nt * 2) * FRAG + lane * 16;
    const unsigned short* sA = sstage + skb0 * 512 + lane * 16;
    unsigned short* qBase = qdst + h * 8 * 512 + (nt * 16 + n16) * 16;
    side_all<0>(aB, wB, sA, sW, qBase);
  };

  // ---- P1: stage mv1/s1 ; P2: left & jleft ----
  stage_mv(mv1 + (size_t)tile * 16384);
  stage_s(s1 + (size_t)tile * 1024, 0);
  __syncthreads();
  run_side(0 * FR_SIDE, OFF_S2MV_F + 0, 0, quads + 0 * 8192);   // left
  run_side(1 * FR_SIDE, OFF_S2MV_F + 4, 0, quads + 1 * 8192);   // jleft
  __syncthreads();
  // ---- P3: stage mv2/s2 ; P4: right & jright ----
  stage_mv(mv2 + (size_t)tile * 16384);
  stage_s(s2 + (size_t)tile * 1024, 2);
  __syncthreads();
  run_side(2 * FR_SIDE, OFF_S2MV_F + 8,  2, quads + 2 * 8192);  // right
  run_side(3 * FR_SIDE, OFF_S2MV_F + 12, 2, quads + 3 * 8192);  // jright
  __syncthreads();

  // ---- P5: geometric product + join (VALU), hidden -> astage (overwrite) ----
#pragma clang loop unroll(disable)
  for (int p = tid; p < 512; p += 64) {
    const int t = p >> 5, cch = p & 31;
    const v16u lu  = *(const v16u*)(quads + 0 * 8192 + (t * 32 + cch) * 16);
    const v16u jlu = *(const v16u*)(quads + 1 * 8192 + (t * 32 + cch) * 16);
    const v16u ru  = *(const v16u*)(quads + 2 * 8192 + (t * 32 + cch) * 16);
    const v16u jru = *(const v16u*)(quads + 3 * 8192 + (t * 32 + cch) * 16);
    float L[16], R[16], JL[16], JR[16];
#pragma unroll
    for (int i = 0; i < 16; ++i) {
      L[i]  = bf2f(lu[i]);  R[i]  = bf2f(ru[i]);
      JL[i] = bf2f(jlu[i]); JR[i] = bf2f(jru[i]);
    }
    float gp[16] = {}, jn[16] = {};
    gpTerms<0, 192>(gp, L, R);
    joinTerms<0, 81>(jn, JL, JR);
    const float refw = refmv[(size_t)(tok0 + t) * 16 + 15];
    const int K = cch & 31;
    const int slot = (K & 7) + 8 * (K >> 4);
    const int ln   = ((K >> 3) & 1) * 16 + t;
#pragma unroll
    for (int i = 0; i < 16; ++i) {
      astage[((i * 2 + 0) * 32 + ln) * 16 + slot] = f2bf(gp[i]);          // hidden ch = cch
      astage[((i * 2 + 1) * 32 + ln) * 16 + slot] = f2bf(jn[i] * refw);   // hidden ch = cch+32
    }
  }
  __syncthreads();   // quads is dead from here on -> reuse as f32 out_mv staging tile

  // ---- P6: output equi-linear (Y=64) + scalar head ----
  float* otile = (float*)quads;                      // 16 tok x 64 ch x 16 comp f32 (64 KB)
  const unsigned short* sA0 = sstage + lane * 16;    // s_cat = [s1 | s2], 4 k-steps
#pragma unroll
  for (int nto = 0; nto < 2; ++nto) {
    const int nt = wave * 2 + nto;
    const unsigned short* wB = ws + (OFF_WOUT_F + nt * 2) * FRAG + lane * 16;
    const unsigned short* sW = ws + (OFF_S2MVO_F + nt * 4) * FRAG + lane * 16;
    float* oBase = otile + h * 8 * 1024 + (nt * 16 + n16) * 16;
    out_all<0>(aB, wB, sA0, sW, oBase);

    // out_s tile for this nt (16 outputs): hidden[...,0] @ mv2s^T + s_cat @ s2s^T + bias
    v8f c = {0.f, 0.f, 0.f, 0.f, 0.f, 0.f, 0.f, 0.f};
#pragma unroll
    for (int kb = 0; kb < 2; ++kb)
      c = wmma_bf16_acc(aB + kb * 512,
                        ws + (OFF_MV2S_F + nt * 2 + kb) * FRAG + lane * 16, c);
#pragma unroll
    for (int skb = 0; skb < 4; ++skb)
      c = wmma_bf16_acc(sA0 + skb * 512,
                        ws + (OFF_S2S_F + nt * 4 + skb) * FRAG + lane * 16, c);
    const int o = nt * 16 + n16;
    const float b = bias[o];
#pragma unroll
    for (int r = 0; r < 8; ++r)
      out_s[(size_t)(tok0 + r + 8 * h) * 64 + o] = c[r] + b;
  }
  __syncthreads();

  // ---- P7: coalesced copy-out of the 64 KB out_mv tile ----
  float* dst = out_mv + (size_t)tile * 16384;
  for (int base = tid * 4; base < 16384; base += 256) {
    const float4 v = *(const float4*)(otile + base);
    *(float4*)(dst + base) = v;
  }
}

// ---------------------------------------------------------------------------
extern "C" void kernel_launch(void* const* d_in, const int* in_sizes, int n_in,
                              void* d_out, int out_size, void* d_ws, size_t ws_size,
                              hipStream_t stream) {
  (void)in_sizes; (void)n_in; (void)out_size; (void)ws_size;
  const float* mv1  = (const float*)d_in[0];
  const float* mv2  = (const float*)d_in[1];
  const float* s1   = (const float*)d_in[2];
  const float* s2   = (const float*)d_in[3];
  const float* ref  = (const float*)d_in[4];
  const float* wL   = (const float*)d_in[5];
  const float* sL   = (const float*)d_in[6];
  const float* wR   = (const float*)d_in[7];
  const float* sR   = (const float*)d_in[8];
  const float* wJL  = (const float*)d_in[9];
  const float* sJL  = (const float*)d_in[10];
  const float* wJR  = (const float*)d_in[11];
  const float* sJR  = (const float*)d_in[12];
  const float* wO   = (const float*)d_in[13];
  const float* sO   = (const float*)d_in[14];
  const float* mv2s = (const float*)d_in[15];
  const float* s2s  = (const float*)d_in[16];
  const float* bias = (const float*)d_in[17];

  unsigned short* ws = (unsigned short*)d_ws;        // needs >= 278528 bytes
  float* out_mv = (float*)d_out;                     // 8*4096*64*16 floats
  float* out_s  = out_mv + (size_t)8 * 4096 * 64 * 16;

  expand_weights<<<(WS_U16 + 255) / 256, 256, 0, stream>>>(
      wL, wJL, wR, wJR, sL, sJL, sR, sJR, wO, sO, mv2s, s2s, ws);
  ga_bilinear_kernel<<<2048, 64, 0, stream>>>(
      mv1, mv2, s1, s2, ref, ws, bias, out_mv, out_s);
}
